// NodeFormerConv_13357348290600
// MI455X (gfx1250) — matrix operand order repaired and checked by reading, not compiled, not run on hardware
//
#include <hip/hip_runtime.h>
#include <hip/hip_bf16.h>
#include <math.h>

#define N_NODES 50000
#define N_EDGES 800000
#define H_HEADS 4
#define D_DIM   64
#define M_FEAT  30
#define K_SAMP  10
#define IN_DIM  128
#define EPS_F   1e-6f

typedef __attribute__((ext_vector_type(16))) __bf16 v16bf;
typedef __attribute__((ext_vector_type(8)))  float  v8f;

union AFrag { v16bf v; unsigned short u[16]; uint4 q[2]; };

__device__ __forceinline__ unsigned short f2bf(float f) {
    __bf16 b = (__bf16)f;                       // hardware f32->bf16 cvt
    return __builtin_bit_cast(unsigned short, b);
}

// order-preserving float<->uint encoding for atomicMax on floats
__device__ __forceinline__ unsigned fenc(float f) {
    unsigned u = __float_as_uint(f);
    return (u & 0x80000000u) ? ~u : (u | 0x80000000u);
}
__device__ __forceinline__ float fdec(unsigned u) {
    unsigned v = (u & 0x80000000u) ? (u & 0x7fffffffu) : ~u;
    return __uint_as_float(v);
}

__device__ __forceinline__ v8f wmma_bf16(const AFrag& a, const AFrag& b, v8f c) {
    return __builtin_amdgcn_wmma_f32_16x16x32_bf16(false, a.v, false, b.v,
                                                   (short)0, c, false, false);
}

// ---------------------------------------------------------------- zero init
__global__ void zero_kernel(float* __restrict__ p, size_t n) {
    size_t t = (size_t)blockIdx.x * blockDim.x + threadIdx.x;
    if (t < n) p[t] = 0.f;
}

// -------------------------- weight pre-pack: bf16, WMMA B-fragment ordered
// B frag layout: lane%16 = col, element i = K (table order: (i>>3)*16 + (lane>>4)*8 + (i&7)).
// wpack[((ct*KKS + kk)*32 + lane)*16 + i]
__global__ void pack_qkv_kernel(const float* __restrict__ Wq,
                                const float* __restrict__ Wk,
                                const float* __restrict__ Wv,
                                unsigned short* __restrict__ wpack) {
    int t = blockIdx.x * blockDim.x + threadIdx.x;   // 48 ct * 4 kk * 32 lanes
    if (t >= 48 * 4 * 32) return;
    int lane = t & 31, kk = (t >> 5) & 3, ct = t >> 7;
    int gcol = ct * 16;
    const float* W;
    if (gcol < 256)      { W = Wq; }
    else if (gcol < 512) { W = Wk; gcol -= 256; }
    else                 { W = Wv; gcol -= 512; }
    int nB = gcol + (lane & 15);
    unsigned short* dst = wpack + (size_t)t * 16;
    #pragma unroll
    for (int i = 0; i < 16; ++i) {
        int kB = kk * 32 + (i >> 3) * 16 + (lane >> 4) * 8 + (i & 7);
        dst[i] = f2bf(W[(size_t)nB * IN_DIM + kB]);
    }
}

__global__ void pack_wo_kernel(const float* __restrict__ Wo,
                               unsigned short* __restrict__ wpack) {
    int t = blockIdx.x * blockDim.x + threadIdx.x;   // 4 ct * 8 kk * 32 lanes
    if (t >= 4 * 8 * 32) return;
    int lane = t & 31, kk = (t >> 5) & 7, ct = t >> 8;
    int nB = ct * 16 + (lane & 15);
    unsigned short* dst = wpack + (size_t)t * 16;
    #pragma unroll
    for (int i = 0; i < 16; ++i) {
        int kB = kk * 32 + (i >> 3) * 16 + (lane >> 4) * 8 + (i & 7);
        dst[i] = f2bf(Wo[(size_t)nB * 256 + kB]);
    }
}

__global__ void pack_proj_kernel(const float* __restrict__ proj,
                                 unsigned short* __restrict__ ppack) {
    int t = blockIdx.x * blockDim.x + threadIdx.x;   // 2 tiles * 2 kk * 32 lanes
    if (t >= 2 * 2 * 32) return;
    int lane = t & 31, kk = (t >> 5) & 1, tile = t >> 6;
    int m = tile * 16 + (lane & 15);
    unsigned short* dst = ppack + (size_t)t * 16;
    #pragma unroll
    for (int i = 0; i < 16; ++i) {
        int kB = kk * 32 + (i >> 3) * 16 + (lane >> 4) * 8 + (i & 7);
        dst[i] = (m < M_FEAT) ? f2bf(proj[(size_t)m * 64 + kB]) : (unsigned short)0;
    }
}

// ------------------------------------------------- K1: fused QKV projection
__global__ __launch_bounds__(128) void qkv_gemm_kernel(
    const float* __restrict__ z, const unsigned short* __restrict__ wpack,
    const float* __restrict__ bq, const float* __restrict__ bk,
    const float* __restrict__ bv,
    float* __restrict__ q_ws, float* __restrict__ k_ws, float* __restrict__ v_ws)
{
    // A tile staged in fragment order: apack[kk][lane][16]
    __shared__ __align__(32) unsigned short apack[4][32][16];
    const int n0 = blockIdx.x * 16;
    const int t  = threadIdx.x;
    for (int idx = t; idx < 4 * 32 * 16; idx += 128) {
        int kk = idx >> 9, ln = (idx >> 4) & 31, i = idx & 15;
        int r = ln & 15, hf = ln >> 4;
        int v8 = i >> 1, g = v8 >> 2, j = v8 & 3;
        int k = kk * 32 + g * 16 + hf * 8 + 2 * j + (i & 1);
        apack[kk][ln][i] = f2bf(z[(size_t)(n0 + r) * IN_DIM + k]);
    }
    __syncthreads();
    const int wave = t >> 5, lane = t & 31;
    const int l15 = lane & 15, half = lane >> 4;
    for (int ct = wave; ct < 48; ct += 4) {
        int gcol = ct * 16;
        const float* bias; float* outp;
        if (gcol < 256)      { bias = bq; outp = q_ws; }
        else if (gcol < 512) { bias = bk; outp = k_ws; gcol -= 256; }
        else                 { bias = bv; outp = v_ws; gcol -= 512; }
        v8f acc = {};
        const unsigned short* wp = wpack + (size_t)ct * 4 * 32 * 16;
        #pragma unroll
        for (int kk = 0; kk < 4; ++kk) {
            AFrag a, b;
            const uint4* ap = (const uint4*)apack[kk][lane];
            a.q[0] = ap[0]; a.q[1] = ap[1];                 // 2x ds_load_b128
            const uint4* bp = (const uint4*)(wp + ((size_t)kk * 32 + lane) * 16);
            __builtin_prefetch(bp + 128, 0, 1);             // global_prefetch_b8
            b.q[0] = bp[0]; b.q[1] = bp[1];                 // 2x global_load_b128
            acc = wmma_bf16(a, b, acc);
        }
        float bcol = bias[gcol + l15];
        #pragma unroll
        for (int j = 0; j < 8; ++j) {
            int row = j + half * 8;                          // C: VGPR j = row, lane = col
            outp[(size_t)(n0 + row) * 256 + gcol + l15] = acc[j] + bcol;
        }
    }
}

// ------------------------------- K2: FAVOR+ feature GEMM + stabilizer + exp
__global__ __launch_bounds__(32) void feat_kernel(
    const float* __restrict__ src, const unsigned short* __restrict__ ppack,
    float* __restrict__ dst, unsigned* __restrict__ kstab, int is_query)
{
    __shared__ __align__(32) unsigned short apack[2][32][16];
    __shared__ float sdiag[16];
    __shared__ unsigned smax[H_HEADS];
    const int r0 = blockIdx.x * 16;
    const int lane = threadIdx.x;
    const float scale = 0.70710678f;       // tau^-0.5 * D^-0.25 = 2 * 64^-0.25
    const float ratio = 0.18257419f;       // M^-0.5
    if (lane < 16) sdiag[lane] = 0.f;
    if (lane < H_HEADS) smax[lane] = 0u;
    __syncthreads();
    {   // pass 1: 0.5*||x||^2 per row (2 lanes per row)
        int row = lane & 15, c0 = (lane >> 4) * 32;
        const float* sr = src + (size_t)(r0 + row) * 64 + c0;
        float s = 0.f;
        #pragma unroll
        for (int c = 0; c < 32; ++c) { float x = sr[c] * scale; s += x * x; }
        atomicAdd(&sdiag[row], 0.5f * s);
    }
    // pass 2: stage scaled bf16 A tile in fragment order
    for (int idx = lane; idx < 2 * 32 * 16; idx += 32) {
        int kk = idx >> 9, ln = (idx >> 4) & 31, i = idx & 15;
        int r = ln & 15, hf = ln >> 4;
        int v8 = i >> 1, g = v8 >> 2, j = v8 & 3;
        int k = kk * 32 + g * 16 + hf * 8 + 2 * j + (i & 1);
        apack[kk][ln][i] = f2bf(src[(size_t)(r0 + r) * 64 + k] * scale);
    }
    __syncthreads();
    const int l15 = lane & 15, half = lane >> 4;
    v8f acc0 = {}, acc1 = {};
    #pragma unroll
    for (int kk = 0; kk < 2; ++kk) {
        AFrag a, b0, b1;
        const uint4* ap = (const uint4*)apack[kk][lane];
        a.q[0] = ap[0]; a.q[1] = ap[1];
        const uint4* p0 = (const uint4*)(ppack + ((size_t)kk * 32 + lane) * 16);
        const uint4* p1 = (const uint4*)(ppack + ((size_t)(2 + kk) * 32 + lane) * 16);
        b0.q[0] = p0[0]; b0.q[1] = p0[1];
        b1.q[0] = p1[0]; b1.q[1] = p1[1];
        acc0 = wmma_bf16(a, b0, acc0);
        acc1 = wmma_bf16(a, b1, acc1);
    }
    const int mcol1 = 16 + l15;
    if (is_query) {
        #pragma unroll
        for (int j = 0; j < 8; ++j) {
            int row = j + half * 8;
            float d0 = acc0[j], d1 = acc1[j];
            float mx = fmaxf(d0, (mcol1 < M_FEAT) ? d1 : -3.4e38f);
            #pragma unroll
            for (int off = 1; off < 16; off <<= 1)   // per-row max across 16 lanes
                mx = fmaxf(mx, __shfl_xor(mx, off, 32));
            float dg = sdiag[row];
            dst[(size_t)(r0 + row) * 32 + l15] = ratio * (expf(d0 - dg - mx) + EPS_F);
            if (mcol1 < M_FEAT)
                dst[(size_t)(r0 + row) * 32 + mcol1] = ratio * (expf(d1 - dg - mx) + EPS_F);
        }
    } else {
        #pragma unroll
        for (int j = 0; j < 8; ++j) {
            int row = j + half * 8;
            int h = (r0 + row) & (H_HEADS - 1);
            float dg = sdiag[row];
            float d0 = acc0[j];
            dst[(size_t)(r0 + row) * 32 + l15] = d0 - dg;   // defer exp: global stab unknown
            atomicMax(&smax[h], fenc(d0));
            if (mcol1 < M_FEAT) {
                float d1 = acc1[j];
                dst[(size_t)(r0 + row) * 32 + mcol1] = d1 - dg;
                atomicMax(&smax[h], fenc(d1));
            }
        }
        __syncthreads();
        if (lane < H_HEADS) atomicMax(&kstab[lane], smax[lane]);
    }
}

// ------------------------------------ K3: finalize kp = exp(dash-diag-stab), ksum
__global__ __launch_bounds__(256) void kp_final_kernel(
    float* __restrict__ kp, const unsigned* __restrict__ kstab,
    float* __restrict__ ksum)
{
    const int h = blockIdx.y;
    const int n0 = blockIdx.x * 64;
    const int t = threadIdx.x;
    const int m = t & 31, sub = t >> 5;
    const float ratio = 0.18257419f;
    const float stab = fdec(kstab[h]);
    __shared__ float part[8][32];
    float s = 0.f;
    for (int i = 0; i < 8; ++i) {
        int n = n0 + sub * 8 + i;
        if (n >= N_NODES) break;
        size_t idx = ((size_t)n * H_HEADS + h) * 32 + m;
        if (m < M_FEAT) {
            float v = ratio * (expf(kp[idx] - stab) + EPS_F);
            kp[idx] = v;
            s += v;
        } else kp[idx] = 0.f;
    }
    part[sub][m] = s;
    __syncthreads();
    if (sub == 0 && m < M_FEAT) {
        float tot = 0.f;
        #pragma unroll
        for (int i = 0; i < 8; ++i) tot += part[i][m];
        atomicAdd(&ksum[h * M_FEAT + m], tot);
    }
}

// ------------------ K4: kvs[h,k,m,d] = sum_n kp*exp(g)*v ; gsum[h,k,m] = sum_n kp*exp(g)
__global__ __launch_bounds__(256) void kvs_kernel(
    const float* __restrict__ kp, const float* __restrict__ v_ws,
    const float* __restrict__ gumbels, float* __restrict__ kvs,
    float* __restrict__ gsum)
{
    const int hk = blockIdx.y;
    const int h = hk / K_SAMP, k = hk % K_SAMP;
    const int n0 = blockIdx.x * 128;
    const int t = threadIdx.x;
    const int d = t & 63, mg = t >> 6;      // 4 m-groups x 8 m's
    float acc[8], gs[8];
    #pragma unroll
    for (int i = 0; i < 8; ++i) { acc[i] = 0.f; gs[i] = 0.f; }
    int nend = (n0 + 128 < N_NODES) ? n0 + 128 : N_NODES;
    for (int n = n0; n < nend; ++n) {
        float vd = v_ws[((size_t)n * H_HEADS + h) * 64 + d];
        float eg = expf(gumbels[((size_t)n * H_HEADS + h) * K_SAMP + k]);
        const float* kprow = kp + ((size_t)n * H_HEADS + h) * 32;
        #pragma unroll
        for (int mm = 0; mm < 8; ++mm) {
            int m = mg + 4 * mm;
            if (m < M_FEAT) {
                float tk = kprow[m] * eg;
                acc[mm] += tk * vd;
                gs[mm]  += tk;
            }
        }
    }
    #pragma unroll
    for (int mm = 0; mm < 8; ++mm) {
        int m = mg + 4 * mm;
        if (m < M_FEAT) {
            atomicAdd(&kvs[(((size_t)h * K_SAMP + k) * M_FEAT + m) * 64 + d], acc[mm]);
            if (d == 0) atomicAdd(&gsum[((size_t)h * K_SAMP + k) * M_FEAT + m], gs[mm]);
        }
    }
}

// ------------------------------------------------------------ degrees
__global__ void deg_kernel(const int* __restrict__ ei,
                           float* __restrict__ din, float* __restrict__ dout_) {
    int e = blockIdx.x * blockDim.x + threadIdx.x;
    if (e >= N_EDGES) return;
    atomicAdd(&dout_[ei[e]], 1.0f);            // row = start
    atomicAdd(&din[ei[N_EDGES + e]], 1.0f);    // col = end
}

// -------------------- K5: z_next = mean_k (qp . kvs) / (qp . gsum)
__global__ __launch_bounds__(64) void attn_kernel(
    const float* __restrict__ qp, const float* __restrict__ kvs,
    const float* __restrict__ gsum, float* __restrict__ znext)
{
    const int n = blockIdx.x, h = blockIdx.y, d = threadIdx.x;
    __shared__ float sq[M_FEAT];
    if (d < M_FEAT) sq[d] = qp[((size_t)n * H_HEADS + h) * 32 + d];
    __syncthreads();
    float acc = 0.f;
    for (int k = 0; k < K_SAMP; ++k) {
        const float* kv = kvs + ((size_t)(h * K_SAMP + k) * M_FEAT) * 64;
        const float* gs = gsum + (size_t)(h * K_SAMP + k) * M_FEAT;
        float den = 0.f, num = 0.f;
        #pragma unroll
        for (int m = 0; m < M_FEAT; ++m) {
            den += sq[m] * gs[m];
            num += sq[m] * kv[(size_t)m * 64 + d];
        }
        acc += num / den;
    }
    znext[((size_t)n * H_HEADS + h) * 64 + d] = acc * (1.0f / K_SAMP);
}

// ---------------- K6: relational bias scatter  z_next += segsum(v[row]*coeff)
__global__ __launch_bounds__(256) void conv_kernel(
    const int* __restrict__ ei, const float* __restrict__ v_ws,
    const float* __restrict__ din, const float* __restrict__ dout_,
    const float* __restrict__ b_rb, float* __restrict__ znext)
{
    __shared__ float sig[H_HEADS];
    if (threadIdx.x < H_HEADS)
        sig[threadIdx.x] = 1.f / (1.f + expf(-b_rb[threadIdx.x]));
    __syncthreads();
    size_t t = (size_t)blockIdx.x * blockDim.x + threadIdx.x;
    if (t >= (size_t)N_EDGES * 64) return;
    int e = (int)(t >> 6), d = (int)(t & 63);
    int row = ei[e], col = ei[N_EDGES + e];
    float norm = rsqrtf(din[col]) * rsqrtf(dout_[row]);
    #pragma unroll
    for (int h = 0; h < H_HEADS; ++h) {
        float msg = v_ws[((size_t)row * H_HEADS + h) * 64 + d] * sig[h] * norm;
        atomicAdd(&znext[((size_t)col * H_HEADS + h) * 64 + d], msg);
    }
}

// ------------------------------------------ K7: output projection (WMMA)
__global__ __launch_bounds__(32) void out_gemm_kernel(
    const float* __restrict__ znext, const unsigned short* __restrict__ wpack,
    const float* __restrict__ bo, float* __restrict__ out)
{
    __shared__ __align__(32) unsigned short apack[8][32][16];
    const int n0 = blockIdx.x * 16;
    const int lane = threadIdx.x;
    for (int idx = lane; idx < 8 * 32 * 16; idx += 32) {
        int kk = idx >> 9, ln = (idx >> 4) & 31, i = idx & 15;
        int r = ln & 15, hf = ln >> 4;
        int v8 = i >> 1, g = v8 >> 2, j = v8 & 3;
        int k = kk * 32 + g * 16 + hf * 8 + 2 * j + (i & 1);
        apack[kk][ln][i] = f2bf(znext[(size_t)(n0 + r) * 256 + k]);
    }
    __syncthreads();
    const int l15 = lane & 15, half = lane >> 4;
    for (int ct = 0; ct < 4; ++ct) {
        v8f acc = {};
        const unsigned short* wp = wpack + (size_t)ct * 8 * 32 * 16;
        #pragma unroll
        for (int kk = 0; kk < 8; ++kk) {
            AFrag a, b;
            const uint4* ap = (const uint4*)apack[kk][lane];
            a.q[0] = ap[0]; a.q[1] = ap[1];
            const uint4* bp = (const uint4*)(wp + ((size_t)kk * 32 + lane) * 16);
            __builtin_prefetch(bp + 128, 0, 1);
            b.q[0] = bp[0]; b.q[1] = bp[1];
            acc = wmma_bf16(a, b, acc);
        }
        float bias = bo[ct * 16 + l15];
        #pragma unroll
        for (int j = 0; j < 8; ++j) {
            int row = j + half * 8;
            out[(size_t)(n0 + row) * 64 + ct * 16 + l15] = acc[j] + bias;
        }
    }
}

// --------------------------------- K8: edge attention weights A_weight[e,h]
__global__ __launch_bounds__(256) void edgew_kernel(
    const int* __restrict__ ei, const float* __restrict__ qp,
    const float* __restrict__ kp, const float* __restrict__ ksum,
    float* __restrict__ aw)
{
    int t = blockIdx.x * blockDim.x + threadIdx.x;
    if (t >= N_EDGES * H_HEADS) return;
    int e = t >> 2, h = t & 3;
    int s = ei[e], en = ei[N_EDGES + e];
    const float* qr = qp + ((size_t)en * H_HEADS + h) * 32;
    const float* kr = kp + ((size_t)s  * H_HEADS + h) * 32;
    const float* ks = ksum + h * M_FEAT;
    float num = 0.f, den = 0.f;
    #pragma unroll
    for (int m = 0; m < M_FEAT; ++m) { num += qr[m] * kr[m]; den += qr[m] * ks[m]; }
    aw[(size_t)e * H_HEADS + h] = num / den;
}

extern "C" void kernel_launch(void* const* d_in, const int* in_sizes, int n_in,
                              void* d_out, int out_size, void* d_ws, size_t ws_size,
                              hipStream_t stream) {
    const float* z    = (const float*)d_in[0];
    const int*   ei   = (const int*)  d_in[1];
    const float* Wq   = (const float*)d_in[2];
    const float* bq   = (const float*)d_in[3];
    const float* Wk   = (const float*)d_in[4];
    const float* bk   = (const float*)d_in[5];
    const float* Wv   = (const float*)d_in[6];
    const float* bv   = (const float*)d_in[7];
    const float* Wo   = (const float*)d_in[8];
    const float* bo   = (const float*)d_in[9];
    const float* b_rb = (const float*)d_in[10];
    const float* proj = (const float*)d_in[11];
    const float* gum  = (const float*)d_in[12];
    float* out = (float*)d_out;

    float* ws    = (float*)d_ws;
    float* q_ws  = ws;
    float* k_ws  = q_ws  + (size_t)N_NODES * 256;
    float* v_ws  = k_ws  + (size_t)N_NODES * 256;
    float* qp    = v_ws  + (size_t)N_NODES * 256;
    float* kp    = qp    + (size_t)N_NODES * H_HEADS * 32;
    float* znext = kp    + (size_t)N_NODES * H_HEADS * 32;
    // contiguous accumulator block (zero-initialized together)
    float* kvs   = znext + (size_t)N_NODES * 256;
    float* gsum  = kvs   + (size_t)H_HEADS * K_SAMP * M_FEAT * 64;
    float* ksum  = gsum  + (size_t)H_HEADS * K_SAMP * M_FEAT;
    unsigned* kstab = (unsigned*)(ksum + H_HEADS * M_FEAT);
    float* din   = (float*)(kstab + H_HEADS);
    float* dout_ = din + N_NODES;
    // packed bf16 weights (32B aligned)
    unsigned short* wqkv_p =
        (unsigned short*)((((uintptr_t)(dout_ + N_NODES)) + 31) & ~(uintptr_t)31);
    unsigned short* wo_p   = wqkv_p + (size_t)48 * 4 * 32 * 16;
    unsigned short* proj_p = wo_p   + (size_t)4 * 8 * 32 * 16;

    size_t zc = (size_t)H_HEADS * K_SAMP * M_FEAT * 64
              + (size_t)H_HEADS * K_SAMP * M_FEAT
              + (size_t)H_HEADS * M_FEAT + H_HEADS + 2 * (size_t)N_NODES;
    zero_kernel<<<(unsigned)((zc + 255) / 256), 256, 0, stream>>>(kvs, zc);

    pack_qkv_kernel<<<(48 * 4 * 32 + 255) / 256, 256, 0, stream>>>(Wq, Wk, Wv, wqkv_p);
    pack_wo_kernel<<<(4 * 8 * 32 + 255) / 256, 256, 0, stream>>>(Wo, wo_p);
    pack_proj_kernel<<<1, 128, 0, stream>>>(proj, proj_p);

    qkv_gemm_kernel<<<N_NODES / 16, 128, 0, stream>>>(z, wqkv_p, bq, bk, bv,
                                                      q_ws, k_ws, v_ws);
    feat_kernel<<<(N_NODES * H_HEADS) / 16, 32, 0, stream>>>(q_ws, proj_p, qp, kstab, 1);
    feat_kernel<<<(N_NODES * H_HEADS) / 16, 32, 0, stream>>>(k_ws, proj_p, kp, kstab, 0);
    kp_final_kernel<<<dim3((N_NODES + 63) / 64, H_HEADS), 256, 0, stream>>>(kp, kstab, ksum);
    kvs_kernel<<<dim3((N_NODES + 127) / 128, H_HEADS * K_SAMP), 256, 0, stream>>>(
        kp, v_ws, gum, kvs, gsum);
    deg_kernel<<<(N_EDGES + 255) / 256, 256, 0, stream>>>(ei, din, dout_);
    attn_kernel<<<dim3(N_NODES, H_HEADS), 64, 0, stream>>>(qp, kvs, gsum, znext);
    conv_kernel<<<(unsigned)(((size_t)N_EDGES * 64 + 255) / 256), 256, 0, stream>>>(
        ei, v_ws, din, dout_, b_rb, znext);
    out_gemm_kernel<<<N_NODES / 16, 32, 0, stream>>>(znext, wo_p, bo, out);
    edgew_kernel<<<(N_EDGES * H_HEADS + 255) / 256, 256, 0, stream>>>(
        ei, qp, kp, ksum, out + (size_t)N_NODES * 64);
}
